// FusedTP3_85779086836287
// MI455X (gfx1250) — compile-verified, matrix operand order checked by reading, not run on hardware
//
#include <hip/hip_runtime.h>

typedef __attribute__((ext_vector_type(16))) _Float16 v16h;
typedef __attribute__((ext_vector_type(8)))  float    v8f;

#define Zdim 8192
#define S0n  16
#define S1n  16
#define SOUTn 32
#define Un   64
#define Pn   256
#define KDIM 256   // k = s0*16 + s1
#define NKT  8     // K tiles of 32
#define NOT2 2     // output tiles of 16

// ---------------------------------------------------------------------------
// Prep: build dense coefficient tensor T[o][k] in LDS via ds atomics, then
// emit WMMA A-fragments (16-bit A 16x32 lane layout) to workspace as f16.
// ---------------------------------------------------------------------------
__global__ __launch_bounds__(256) void tp_prep(const float* __restrict__ coef,
                                               const int*   __restrict__ idx,
                                               _Float16*    __restrict__ Aws) {
    __shared__ float Tf[SOUTn * KDIM];            // 32 KB
    const int tid = threadIdx.x;
    for (int i = tid; i < SOUTn * KDIM; i += 256) Tf[i] = 0.f;
    __syncthreads();
    {   // one path per thread (P == 256 == blockDim)
        const int p  = tid;
        const int i0 = idx[3 * p + 0];
        const int i1 = idx[3 * p + 1];
        const int io = idx[3 * p + 2];
        atomicAdd(&Tf[io * KDIM + i0 * 16 + i1], coef[p]);
    }
    __syncthreads();
    // A-fragment layout: element e of lane l, tile = otile*8 + ktile
    //   K = ktile*32 + (e<8 ? e : e+8) + 8*(l>=16),  M = otile*16 + (l&15)
    for (int t = tid; t < NOT2 * NKT * 32 * 16; t += 256) {
        const int e     = t & 15;
        const int lane  = (t >> 4) & 31;
        const int tile  = t >> 9;
        const int otile = tile >> 3;
        const int ktile = tile & 7;
        const int hi    = lane >> 4;
        const int base  = (e < 8) ? e : (e + 8);
        const int k     = ktile * 32 + base + 8 * hi;
        const int o     = otile * 16 + (lane & 15);
        Aws[t] = (_Float16)Tf[o * KDIM + k];
    }
}

// ---------------------------------------------------------------------------
// Main: each wave handles 16 column-tiles (16 consecutive u of one z row per
// tile).  Per tile: load 16 a + 8 b floats per lane, build outer-product B
// fragments, chain 16 v_wmma_f32_16x16x32_f16 (2 o-tiles x 8 K-tiles),
// store 32 output rows.  Uniform control flow -> EXEC all ones for WMMA.
// ---------------------------------------------------------------------------
__global__ __launch_bounds__(256) void tp_main(const float*    __restrict__ x0,
                                               const float*    __restrict__ x1,
                                               const _Float16* __restrict__ Aws,
                                               float*          __restrict__ out) {
    __shared__ alignas(32) _Float16 Alds[NOT2 * NKT * 32 * 16];   // 16 KB
    const int tid = threadIdx.x;
    // stage A-fragments to LDS (16B vector copies)
    for (int i = tid; i < (NOT2 * NKT * 32 * 16) / 8; i += 256)
        ((uint4*)Alds)[i] = ((const uint4*)Aws)[i];
    __syncthreads();

    const int lane = tid & 31;
    const int n    = lane & 15;
    const int hi   = lane >> 4;
    const int w    = blockIdx.x * 8 + (tid >> 5);     // global wave id (2048)

    for (int i = 0; i < 16; ++i) {                    // 16 tiles per wave
        const int t  = w * 16 + i;                    // 32768 tiles total
        const int z  = t >> 2;
        const int u0 = (t & 3) * 16;
        const float* b0 = x0 + z * (S0n * Un);
        const float* b1 = x1 + z * (S1n * Un);
        const int col = u0 + n;

        if (i + 1 < 16) {                             // gfx1250 global_prefetch_b8
            const int t2 = t + 1, z2 = t2 >> 2, c2 = (t2 & 3) * 16 + n;
            __builtin_prefetch(x0 + z2 * (S0n * Un) + c2, 0, 0);
            __builtin_prefetch(x1 + z2 * (S1n * Un) + c2, 0, 0);
        }

        float a[16];
        #pragma unroll
        for (int s = 0; s < 16; ++s) a[s] = b0[s * Un + col];
        float bb[8];
        #pragma unroll
        for (int j = 0; j < 8; ++j)  bb[j] = b1[(hi * 8 + j) * Un + col];

        v8f d0 = {}; v8f d1 = {};
        #pragma unroll
        for (int kt = 0; kt < NKT; ++kt) {
            v16h B;
            #pragma unroll
            for (int e = 0; e < 16; ++e)
                B[e] = (_Float16)(a[2 * kt + (e >> 3)] * bb[e & 7]);
            const v16h A0 = *(const v16h*)(Alds + ((0 * NKT + kt) * 32 + lane) * 16);
            const v16h A1 = *(const v16h*)(Alds + ((1 * NKT + kt) * 32 + lane) * 16);
            d0 = __builtin_amdgcn_wmma_f32_16x16x32_f16(false, A0, false, B,
                                                        (short)0, d0, false, false);
            d1 = __builtin_amdgcn_wmma_f32_16x16x32_f16(false, A1, false, B,
                                                        (short)0, d1, false, false);
        }

        float* op = out + z * (SOUTn * Un) + u0 + n;
        #pragma unroll
        for (int j = 0; j < 8; ++j) {
            op[(8 * hi + j) * Un]        = d0[j];   // o-tile 0: rows 0..15
            op[(16 + 8 * hi + j) * Un]   = d1[j];   // o-tile 1: rows 16..31
        }
    }
}

extern "C" void kernel_launch(void* const* d_in, const int* in_sizes, int n_in,
                              void* d_out, int out_size, void* d_ws, size_t ws_size,
                              hipStream_t stream) {
    const float* x0   = (const float*)d_in[0];
    const float* x1   = (const float*)d_in[1];
    const float* coef = (const float*)d_in[2];
    const int*   idx  = (const int*)d_in[3];
    _Float16* Aws = (_Float16*)d_ws;                 // 16 KB of workspace

    tp_prep<<<1, 256, 0, stream>>>(coef, idx, Aws);
    tp_main<<<256, 256, 0, stream>>>(x0, x1, Aws, (float*)d_out);
}
